// GNNModel_53334903881954
// MI455X (gfx1250) — matrix operand order repaired and checked by reading, not compiled, not run on hardware
//
#include <hip/hip_runtime.h>
#include <math.h>

#define D 128
#define H 128
#define G 64
#define C 2

typedef __attribute__((ext_vector_type(16))) _Float16 v16h;
typedef __attribute__((ext_vector_type(8)))  _Float16 v8h;
typedef __attribute__((ext_vector_type(8)))  float    v8f;

__device__ __forceinline__ void atomic_add_f32(float* p, float v) {
  __hip_atomic_fetch_add(p, v, __ATOMIC_RELAXED, __HIP_MEMORY_SCOPE_AGENT);
}

__device__ __forceinline__ v16h join16(v8h lo, v8h hi) {
  v16h r;
  #pragma unroll
  for (int i = 0; i < 8; ++i) { r[i] = lo[i]; r[i + 8] = hi[i]; }
  return r;
}

// ---------------------------------------------------------------------------
// GEMM: Out[N,128] = X[N,128] @ W[128,128], f16 WMMA with f32 accumulate.
// Block = 256 threads = 8 waves; wave w computes rows [blk*128 + w*16, +16).
// Per K-step: preload all 8 B fragments, then 8 back-to-back WMMAs.
// ---------------------------------------------------------------------------
__global__ __launch_bounds__(256)
void gemm_xw_f16(const float* __restrict__ X, const float* __restrict__ W,
                 float* __restrict__ Out, int nrows)
{
  __shared__ _Float16 ldsW[H * H];       // transposed: ldsW[j*H + k] = W[k][j]
  __shared__ _Float16 ldsX[8][16 * D];   // per-wave 16-row tile, f16, row-major

  const int tid  = threadIdx.x;
  const int wave = tid >> 5;
  const int lane = tid & 31;
  const int grp  = lane >> 4;   // 0 or 1 (half-wave)
  const int mn   = lane & 15;   // row (A) / col (B,D) index within tile

  // Stage W (fp32 row-major [k][j]) into LDS transposed as f16.
  for (int i = tid; i < H * H; i += 256) {
    const int k = i >> 7, j = i & (H - 1);
    ldsW[j * H + k] = (_Float16)W[i];
  }

  // Stage this wave's 16 rows of X as f16 (zero-pad OOB rows; keeps EXEC full
  // through the WMMA region).
  const int row0 = blockIdx.x * 128 + wave * 16;
  for (int it = 0; it < 64; ++it) {
    const int idx = it * 32 + lane;          // 0 .. 2047
    const int r = idx >> 7, c = idx & (D - 1);
    const int row = row0 + r;
    ldsX[wave][idx] = (row < nrows) ? (_Float16)X[row * D + c] : (_Float16)0.0f;
  }
  __syncthreads();

  v8f acc[8] = {};

  #pragma unroll
  for (int kt = 0; kt < 4; ++kt) {
    // A fragment (16x32 f16): lanes 0-15 = rows, halves 0..7 -> k = kt*32+grp*8+h,
    // halves 8..15 -> k = kt*32+16+grp*8+(h-8). Two ds_load_b128 per lane.
    const _Float16* ap = &ldsX[wave][mn * D + kt * 32 + grp * 8];
    v16h a = join16(*(const v8h*)ap, *(const v8h*)(ap + 16));

    // Preload all 8 B fragments for this K-step (one deep LDS clause),
    // then issue the 8 WMMAs back-to-back.
    v16h bfrag[8];
    #pragma unroll
    for (int jt = 0; jt < 8; ++jt) {
      // B fragment (32x16 f16): lane col n = mn, same k order as A
      // (halves 0..7 -> k = kt*32+grp*8+h, halves 8..15 -> k = kt*32+16+grp*8+(h-8)).
      const _Float16* bp = &ldsW[(jt * 16 + mn) * H + kt * 32 + grp * 8];
      bfrag[jt] = join16(*(const v8h*)bp, *(const v8h*)(bp + 16));
    }
    #pragma unroll
    for (int jt = 0; jt < 8; ++jt) {
      acc[jt] = __builtin_amdgcn_wmma_f32_16x16x32_f16(
          false, a, false, bfrag[jt], (short)0, acc[jt], false, false);
    }
  }

  // D tile layout: VGPR r holds row m = r + 8*grp, col = jt*16 + mn.
  #pragma unroll
  for (int jt = 0; jt < 8; ++jt) {
    #pragma unroll
    for (int r = 0; r < 8; ++r) {
      const int row = row0 + r + 8 * grp;
      if (row < nrows) Out[row * H + jt * 16 + mn] = acc[jt][r];
    }
  }
}

// ---------------------------------------------------------------------------
// Degree / normalization
// ---------------------------------------------------------------------------
__global__ void deg_init(float* deg, int n) {
  int i = blockIdx.x * blockDim.x + threadIdx.x;
  if (i < n) deg[i] = 1.0f;                      // self-loop
}
__global__ void deg_edges(const int* __restrict__ dst, float* deg, int e) {
  int i = blockIdx.x * blockDim.x + threadIdx.x;
  if (i < e) atomic_add_f32(&deg[dst[i]], 1.0f);
}
__global__ void deg_to_dinv(float* deg, int n) {
  int i = blockIdx.x * blockDim.x + threadIdx.x;
  if (i < n) deg[i] = rsqrtf(deg[i]);
}

// agg = h * dinv^2 + bias   (also serves as the zero-init of the accumulator)
__global__ void self_init(const float* __restrict__ h, const float* __restrict__ dinv,
                          const float* __restrict__ bias, float* __restrict__ agg,
                          int nelem) {
  int idx = blockIdx.x * blockDim.x + threadIdx.x;
  if (idx < nelem) {
    int i = idx >> 7, c = idx & (H - 1);
    float di = dinv[i];
    agg[idx] = h[idx] * di * di + bias[c];
  }
}

// One wave per edge; lane covers 4 channels (float4 gather + 4 f32 atomics).
__global__ __launch_bounds__(256)
void edge_scatter(const int* __restrict__ src, const int* __restrict__ dst,
                  const float* __restrict__ dinv, const float* __restrict__ h,
                  float* __restrict__ agg, int e)
{
  int gw   = (blockIdx.x * 256 + threadIdx.x) >> 5;   // global wave id = edge id
  int lane = threadIdx.x & 31;
  if (gw >= e) return;
  int s = src[gw], d = dst[gw];
  float coef = dinv[s] * dinv[d];
  const float4 hv = *(const float4*)(h + s * H + lane * 4);
  float* base = agg + d * H + lane * 4;
  atomic_add_f32(base + 0, hv.x * coef);
  atomic_add_f32(base + 1, hv.y * coef);
  atomic_add_f32(base + 2, hv.z * coef);
  atomic_add_f32(base + 3, hv.w * coef);
}

__global__ void relu_ip(float* x, int n) {
  int i = blockIdx.x * blockDim.x + threadIdx.x;
  if (i < n) x[i] = fmaxf(x[i], 0.0f);
}

// ---------------------------------------------------------------------------
// Pooling + classifier
// ---------------------------------------------------------------------------
__global__ void zero_f(float* p, int n) {
  int i = blockIdx.x * blockDim.x + threadIdx.x;
  if (i < n) p[i] = 0.0f;
}
__global__ void pool_cnt(const int* __restrict__ batch, float* cnt, int n) {
  int i = blockIdx.x * blockDim.x + threadIdx.x;
  if (i < n) atomic_add_f32(&cnt[batch[i]], 1.0f);
}
// batch is SORTED: accumulate runs of identical graph id in registers and
// flush once per run -> ~16x fewer atomics on the 64x128 sum buffer.
// Block = 256 threads: (channel c = tid&127, half = tid>>7); each thread
// walks 32 consecutive rows for its channel.
__global__ __launch_bounds__(256)
void pool_sum_sorted(const float* __restrict__ x, const int* __restrict__ batch,
                     float* sums, int nrows) {
  const int c    = threadIdx.x & (H - 1);
  const int half = threadIdx.x >> 7;              // 0 or 1
  const int base = blockIdx.x * 64 + half * 32;
  float acc = 0.0f;
  int curg = -1;
  for (int r = 0; r < 32; ++r) {
    const int row = base + r;
    if (row >= nrows) break;
    const int g = batch[row];
    if (g != curg) {
      if (curg >= 0) atomic_add_f32(&sums[curg * H + c], acc);
      curg = g;
      acc = 0.0f;
    }
    acc += x[row * H + c];
  }
  if (curg >= 0) atomic_add_f32(&sums[curg * H + c], acc);
}
__global__ void pool_div(const float* __restrict__ sums, const float* __restrict__ cnt,
                         float* pooled, int n) {
  int idx = blockIdx.x * blockDim.x + threadIdx.x;
  if (idx < n) {
    int g = idx >> 7;
    pooled[idx] = sums[idx] / fmaxf(cnt[g], 1.0f);
  }
}
// hcls = relu([p,p] @ Wc1 + bc1) = relu(p @ (Wc1_top + Wc1_bot) + bc1)
__global__ void cls1(const float* __restrict__ pooled, const float* __restrict__ Wc1,
                     const float* __restrict__ bc1, float* hcls, int n) {
  int idx = blockIdx.x * blockDim.x + threadIdx.x;
  if (idx < n) {
    int g = idx >> 7, j = idx & (H - 1);
    float acc = bc1[j];
    #pragma unroll 4
    for (int k = 0; k < H; ++k)
      acc += pooled[g * H + k] * (Wc1[k * H + j] + Wc1[(k + H) * H + j]);
    hcls[idx] = fmaxf(acc, 0.0f);
  }
}
__global__ void cls2(const float* __restrict__ hcls, const float* __restrict__ Wc2,
                     const float* __restrict__ bc2, float* out, int n) {
  int idx = blockIdx.x * blockDim.x + threadIdx.x;
  if (idx < n) {
    int g = idx >> 1, c = idx & 1;
    float acc = bc2[c];
    #pragma unroll 4
    for (int j = 0; j < H; ++j)
      acc += hcls[g * H + j] * Wc2[j * C + c];
    out[idx] = acc;
  }
}

// ---------------------------------------------------------------------------
extern "C" void kernel_launch(void* const* d_in, const int* in_sizes, int n_in,
                              void* d_out, int out_size, void* d_ws, size_t ws_size,
                              hipStream_t stream)
{
  const float* x     = (const float*)d_in[0];
  const int*   ei    = (const int*)  d_in[1];
  const int*   batch = (const int*)  d_in[2];
  const float* Wl[3] = {(const float*)d_in[3], (const float*)d_in[5], (const float*)d_in[7]};
  const float* bl[3] = {(const float*)d_in[4], (const float*)d_in[6], (const float*)d_in[8]};
  const float* Wc1   = (const float*)d_in[9];
  const float* bc1   = (const float*)d_in[10];
  const float* Wc2   = (const float*)d_in[11];
  const float* bc2   = (const float*)d_in[12];
  float* out = (float*)d_out;
  (void)n_in; (void)out_size; (void)ws_size;

  const int N = in_sizes[0] / D;
  const int E = in_sizes[1] / 2;
  const int* src = ei;
  const int* dst = ei + E;
  const int nelem = N * H;

  // Workspace carve-up (all buffers fully written before being read).
  char* ws = (char*)d_ws;
  size_t off = 0;
  auto carve = [&](size_t bytes) -> void* {
    void* p = ws + off;
    off = (off + bytes + 255) & ~(size_t)255;
    return p;
  };
  float* dinv   = (float*)carve((size_t)N * sizeof(float));
  float* bufA   = (float*)carve((size_t)nelem * sizeof(float));   // agg / next x
  float* bufB   = (float*)carve((size_t)nelem * sizeof(float));   // h = x @ W
  float* sums   = (float*)carve((size_t)G * H * sizeof(float));
  float* cnt    = (float*)carve((size_t)G * sizeof(float));
  float* pooled = (float*)carve((size_t)G * H * sizeof(float));
  float* hcls   = (float*)carve((size_t)G * H * sizeof(float));

  // Degree -> dinv
  deg_init   <<<(N + 255) / 256, 256, 0, stream>>>(dinv, N);
  deg_edges  <<<(E + 255) / 256, 256, 0, stream>>>(dst, dinv, E);
  deg_to_dinv<<<(N + 255) / 256, 256, 0, stream>>>(dinv, N);

  // 3 GCN layers
  const float* xin = x;
  for (int l = 0; l < 3; ++l) {
    gemm_xw_f16 <<<(N + 127) / 128, 256, 0, stream>>>(xin, Wl[l], bufB, N);
    self_init   <<<(nelem + 255) / 256, 256, 0, stream>>>(bufB, dinv, bl[l], bufA, nelem);
    edge_scatter<<<(E + 7) / 8, 256, 0, stream>>>(src, dst, dinv, bufB, bufA, E);
    relu_ip     <<<(nelem + 255) / 256, 256, 0, stream>>>(bufA, nelem);
    xin = bufA;
  }

  // Global mean pool + classifier
  zero_f         <<<(G * H + 255) / 256, 256, 0, stream>>>(sums, G * H);
  zero_f         <<<1, 256, 0, stream>>>(cnt, G);
  pool_cnt       <<<(N + 255) / 256, 256, 0, stream>>>(batch, cnt, N);
  pool_sum_sorted<<<(N + 63) / 64, 256, 0, stream>>>(bufA, batch, sums, N);
  pool_div       <<<(G * H + 255) / 256, 256, 0, stream>>>(sums, cnt, pooled, G * H);
  cls1           <<<(G * H + 255) / 256, 256, 0, stream>>>(pooled, Wc1, bc1, hcls, G * H);
  cls2           <<<1, 128, 0, stream>>>(hcls, Wc2, bc2, out, G * C);
}
